// MixerBlock_40982577938925
// MI455X (gfx1250) — compile-verified, matrix operand order checked by reading, not run on hardware
//
#include <hip/hip_runtime.h>
#include <hip/hip_bf16.h>

#define NND  50000
#define NEDG 400000
#define NP   50048   // N padded to multiple of 32 for split-K GEMMs

typedef __attribute__((ext_vector_type(16))) __bf16 bf16x16;
typedef __attribute__((ext_vector_type(8)))  float  f32x8;

__device__ __forceinline__ f32x8 wmma_bf16(bf16x16 a, bf16x16 b, f32x8 c) {
  return __builtin_amdgcn_wmma_f32_16x16x32_bf16(false, a, false, b, (short)0, c, false, false);
}

__device__ __forceinline__ f32x8 zero8() {
  f32x8 z;
#pragma unroll
  for (int i = 0; i < 8; ++i) z[i] = 0.f;
  return z;
}

__device__ __forceinline__ __bf16 f2bf(float f) {
  union { float f; unsigned u; } x; x.f = f;
  unsigned r = (x.u + 0x7FFFu + ((x.u >> 16) & 1u)) >> 16;
  unsigned short us = (unsigned short)r;
  __bf16 out; __builtin_memcpy(&out, &us, 2); return out;
}

__device__ __forceinline__ float silu(float x) { return x / (1.f + __expf(-x)); }

// ---------------- prep kernels ----------------

// W [din,dout] fp32  ->  Wt [dout][din] bf16 (K contiguous per output column)
__global__ void k_w_t(const float* __restrict__ W, __bf16* __restrict__ Wt,
                      int din, int dout) {
  int i = blockIdx.x * blockDim.x + threadIdx.x;
  if (i >= din * dout) return;
  int k = i / dout, n = i % dout;
  Wt[n * din + k] = f2bf(W[i]);
}

__global__ void k_f2bf(const float* __restrict__ x, __bf16* __restrict__ y, long n) {
  long i = (long)blockIdx.x * blockDim.x + threadIdx.x;
  if (i < n) y[i] = f2bf(x[i]);
}

// V' = V + cond ; Vin = bf16(concat(V', s_enc)) [N,160]; Vp = V' fp32
__global__ void k_prep_nodes(const float* __restrict__ V, const float* __restrict__ cond,
                             const float* __restrict__ senc,
                             float* __restrict__ Vp, __bf16* __restrict__ Vin) {
  long i = (long)blockIdx.x * blockDim.x + threadIdx.x;
  if (i >= (long)NND * 160) return;
  int n = (int)(i / 160), c = (int)(i % 160);
  float v;
  if (c < 128) { v = V[(long)n * 128 + c] + cond[(long)n * 128 + c]; Vp[(long)n * 128 + c] = v; }
  else         { v = senc[(long)n * 32 + (c - 128)]; }
  Vin[i] = f2bf(v);
}

// ---------------- fused edge MLP: gather -> GEMM1 -> SiLU -> GEMM2 -> LN -> residual + scatter ----
__global__ __launch_bounds__(128) void k_edge(
    const __bf16* __restrict__ Ebf, const __bf16* __restrict__ Vin,
    const int* __restrict__ src, const int* __restrict__ dst,
    const __bf16* __restrict__ W1t, const float* __restrict__ b1,
    const __bf16* __restrict__ W2t, const float* __restrict__ b2,
    const float* __restrict__ g, const float* __restrict__ bta,
    const float* __restrict__ Ein, float* __restrict__ Eout, float* __restrict__ agg) {
  __shared__ __attribute__((aligned(32))) __bf16 Hl[64][128];
  __shared__ __attribute__((aligned(32))) float  Sl[64][128];
  int tid = threadIdx.x;
  int w = tid >> 5, lane = tid & 31;
  int r16 = lane & 15, kh = (lane >> 4) << 4, mh = (lane >> 4) << 3;
  int ebase = (blockIdx.x * 4 + w) * 16;
  int e = ebase + r16;
  int es = src[e], er = dst[e];
  __builtin_prefetch(W1t, 0, 1);

  f32x8 acc[8];
#pragma unroll
  for (int t = 0; t < 8; ++t) acc[t] = zero8();
#pragma unroll
  for (int kt = 0; kt < 14; ++kt) {
    int col = kt * 32 + kh;
    const __bf16* ap;
    if (col < 128)      ap = Ebf + (long)e * 128 + col;
    else if (col < 288) ap = Vin + (long)es * 160 + (col - 128);
    else                ap = Vin + (long)er * 160 + (col - 288);
    bf16x16 a = *(const bf16x16*)ap;
#pragma unroll
    for (int nt = 0; nt < 8; ++nt) {
      bf16x16 b = *(const bf16x16*)(W1t + (nt * 16 + r16) * 448 + kt * 32 + kh);
      acc[nt] = wmma_bf16(a, b, acc[nt]);
    }
  }
#pragma unroll
  for (int nt = 0; nt < 8; ++nt) {
    float bb = b1[nt * 16 + r16];
#pragma unroll
    for (int v = 0; v < 8; ++v)
      Hl[w * 16 + v + mh][nt * 16 + r16] = f2bf(silu(acc[nt][v] + bb));
  }
  __syncthreads();

  f32x8 acc2[8];
#pragma unroll
  for (int t = 0; t < 8; ++t) acc2[t] = zero8();
#pragma unroll
  for (int kt = 0; kt < 4; ++kt) {
    bf16x16 a = *(const bf16x16*)&Hl[w * 16 + r16][kt * 32 + kh];
#pragma unroll
    for (int nt = 0; nt < 8; ++nt) {
      bf16x16 b = *(const bf16x16*)(W2t + (nt * 16 + r16) * 128 + kt * 32 + kh);
      acc2[nt] = wmma_bf16(a, b, acc2[nt]);
    }
  }
#pragma unroll
  for (int nt = 0; nt < 8; ++nt) {
    float bb = b2[nt * 16 + r16];
#pragma unroll
    for (int v = 0; v < 8; ++v)
      Sl[w * 16 + v + mh][nt * 16 + r16] = acc2[nt][v] + bb;
  }
  __syncthreads();

  if (tid < 64) {
    int ee = blockIdx.x * 64 + tid;
    float mean = 0.f;
    for (int c = 0; c < 128; ++c) mean += Sl[tid][c];
    mean *= (1.f / 128.f);
    float var = 0.f;
    for (int c = 0; c < 128; ++c) { float d = Sl[tid][c] - mean; var += d * d; }
    float inv = rsqrtf(var * (1.f / 128.f) + 1e-5f);
    int rr = dst[ee];
    for (int c = 0; c < 128; ++c) {
      float y = (Sl[tid][c] - mean) * inv * g[c] + bta[c];
      Eout[(long)ee * 128 + c] = Ein[(long)ee * 128 + c] + y;
      atomicAdd(&agg[(long)rr * 128 + c], y);
    }
  }
}

// ---------------- fused node MLP -> LN -> residual ----------------
__global__ __launch_bounds__(128) void k_node(
    const __bf16* __restrict__ Vin, const __bf16* __restrict__ aggbf,
    const __bf16* __restrict__ W1t, const float* __restrict__ b1,
    const __bf16* __restrict__ W2t, const float* __restrict__ b2,
    const float* __restrict__ g, const float* __restrict__ bta,
    const float* __restrict__ Vp, float* __restrict__ V1) {
  __shared__ __attribute__((aligned(32))) __bf16 Hl[64][128];
  __shared__ __attribute__((aligned(32))) float  Sl[64][128];
  int tid = threadIdx.x;
  int w = tid >> 5, lane = tid & 31;
  int r16 = lane & 15, kh = (lane >> 4) << 4, mh = (lane >> 4) << 3;
  int nbase = (blockIdx.x * 4 + w) * 16;
  int node = nbase + r16; if (node >= NND) node = NND - 1;

  f32x8 acc[8];
#pragma unroll
  for (int t = 0; t < 8; ++t) acc[t] = zero8();
#pragma unroll
  for (int kt = 0; kt < 9; ++kt) {
    int col = kt * 32 + kh;
    const __bf16* ap = (col < 160) ? (Vin + (long)node * 160 + col)
                                   : (aggbf + (long)node * 128 + (col - 160));
    bf16x16 a = *(const bf16x16*)ap;
#pragma unroll
    for (int nt = 0; nt < 8; ++nt) {
      bf16x16 b = *(const bf16x16*)(W1t + (nt * 16 + r16) * 288 + kt * 32 + kh);
      acc[nt] = wmma_bf16(a, b, acc[nt]);
    }
  }
#pragma unroll
  for (int nt = 0; nt < 8; ++nt) {
    float bb = b1[nt * 16 + r16];
#pragma unroll
    for (int v = 0; v < 8; ++v)
      Hl[w * 16 + v + mh][nt * 16 + r16] = f2bf(silu(acc[nt][v] + bb));
  }
  __syncthreads();

  f32x8 acc2[8];
#pragma unroll
  for (int t = 0; t < 8; ++t) acc2[t] = zero8();
#pragma unroll
  for (int kt = 0; kt < 4; ++kt) {
    bf16x16 a = *(const bf16x16*)&Hl[w * 16 + r16][kt * 32 + kh];
#pragma unroll
    for (int nt = 0; nt < 8; ++nt) {
      bf16x16 b = *(const bf16x16*)(W2t + (nt * 16 + r16) * 128 + kt * 32 + kh);
      acc2[nt] = wmma_bf16(a, b, acc2[nt]);
    }
  }
#pragma unroll
  for (int nt = 0; nt < 8; ++nt) {
    float bb = b2[nt * 16 + r16];
#pragma unroll
    for (int v = 0; v < 8; ++v)
      Sl[w * 16 + v + mh][nt * 16 + r16] = acc2[nt][v] + bb;
  }
  __syncthreads();

  int row = blockIdx.x * 64 + tid;
  if (tid < 64 && row < NND) {
    float mean = 0.f;
    for (int c = 0; c < 128; ++c) mean += Sl[tid][c];
    mean *= (1.f / 128.f);
    float var = 0.f;
    for (int c = 0; c < 128; ++c) { float d = Sl[tid][c] - mean; var += d * d; }
    float inv = rsqrtf(var * (1.f / 128.f) + 1e-5f);
    for (int c = 0; c < 128; ++c) {
      float y = (Sl[tid][c] - mean) * inv * g[c] + bta[c];
      V1[(long)row * 128 + c] = Vp[(long)row * 128 + c] + y;
    }
  }
}

// ---------------- LayerNorm rows (wave per row) -> bf16 out ----------------
__global__ void k_ln_rows(const float* __restrict__ X, const float* __restrict__ g,
                          const float* __restrict__ b, __bf16* __restrict__ out, int nrows) {
  int gid = blockIdx.x * blockDim.x + threadIdx.x;
  int wid = gid >> 5, lane = gid & 31;
  if (wid >= nrows) return;
  float4 v = ((const float4*)(X + (long)wid * 128))[lane];
  float s = v.x + v.y + v.z + v.w;
#pragma unroll
  for (int m = 16; m >= 1; m >>= 1) s += __shfl_xor(s, m);
  float mean = s * (1.f / 128.f);
  float dx = v.x - mean, dy = v.y - mean, dz = v.z - mean, dw = v.w - mean;
  float q = dx * dx + dy * dy + dz * dz + dw * dw;
#pragma unroll
  for (int m = 16; m >= 1; m >>= 1) q += __shfl_xor(q, m);
  float inv = rsqrtf(q * (1.f / 128.f) + 1e-5f);
  int c = lane * 4;
  long o = (long)wid * 128 + c;
  out[o + 0] = f2bf(dx * inv * g[c + 0] + b[c + 0]);
  out[o + 1] = f2bf(dy * inv * g[c + 1] + b[c + 1]);
  out[o + 2] = f2bf(dz * inv * g[c + 2] + b[c + 2]);
  out[o + 3] = f2bf(dw * inv * g[c + 3] + b[c + 3]);
}

// ---------------- xm / fx projections (fx stored transposed [128][NP]) ----------------
__global__ __launch_bounds__(128) void k_proj(
    const __bf16* __restrict__ x, const __bf16* __restrict__ Wxt,
    const __bf16* __restrict__ Wfxt, const float* __restrict__ bx,
    const float* __restrict__ bfx, __bf16* __restrict__ xm, __bf16* __restrict__ fxT) {
  int tid = threadIdx.x;
  int w = tid >> 5, lane = tid & 31;
  int r16 = lane & 15, kh = (lane >> 4) << 4, mh = (lane >> 4) << 3;
  int nbase = (blockIdx.x * 4 + w) * 16;
  int row = nbase + r16; if (row >= NND) row = NND - 1;
#pragma unroll
  for (int which = 0; which < 2; ++which) {
    const __bf16* Wt = which ? Wfxt : Wxt;
    const float* bias = which ? bfx : bx;
    f32x8 acc[8];
#pragma unroll
    for (int t = 0; t < 8; ++t) acc[t] = zero8();
#pragma unroll
    for (int kt = 0; kt < 4; ++kt) {
      bf16x16 a = *(const bf16x16*)(x + (long)row * 128 + kt * 32 + kh);
#pragma unroll
      for (int nt = 0; nt < 8; ++nt) {
        bf16x16 b = *(const bf16x16*)(Wt + (nt * 16 + r16) * 128 + kt * 32 + kh);
        acc[nt] = wmma_bf16(a, b, acc[nt]);
      }
    }
#pragma unroll
    for (int nt = 0; nt < 8; ++nt) {
      int n = nt * 16 + r16;
      float bb = bias[n];
#pragma unroll
      for (int v = 0; v < 8; ++v) {
        int orow = nbase + v + mh;
        if (orow < NND) {
          float val = acc[nt][v] + bb;
          if (which == 0) xm[(long)orow * 128 + n] = f2bf(val);
          else            fxT[(long)n * NP + orow] = f2bf(val);
        }
      }
    }
  }
}

// ---------------- slice logits + softmax over M=64; write sw [n][256] and swT [256][NP] -----
__global__ __launch_bounds__(128) void k_slice(
    const __bf16* __restrict__ xm, const __bf16* __restrict__ WsT,
    const float* __restrict__ temp, __bf16* __restrict__ sw_nm,
    __bf16* __restrict__ swT, float* __restrict__ normg) {
  __shared__ float Sl[4][16][64];
  __shared__ float nrm[256];
  int tid = threadIdx.x;
  int w = tid >> 5, lane = tid & 31;
  int r16 = lane & 15, kh = (lane >> 4) << 4, mh = (lane >> 4) << 3;
  for (int i = tid; i < 256; i += 128) nrm[i] = 0.f;
  __syncthreads();
  int nbase = (blockIdx.x * 4 + w) * 16;
  int rowc = nbase + r16; if (rowc >= NND) rowc = NND - 1;
  float invt = 1.f / temp[0];
  for (int h = 0; h < 4; ++h) {
    f32x8 acc[4];
#pragma unroll
    for (int t = 0; t < 4; ++t) acc[t] = zero8();
    bf16x16 a = *(const bf16x16*)(xm + (long)rowc * 128 + h * 32 + kh);
#pragma unroll
    for (int mt = 0; mt < 4; ++mt) {
      bf16x16 b = *(const bf16x16*)(WsT + (mt * 16 + r16) * 32 + kh);
      acc[mt] = wmma_bf16(a, b, acc[mt]);
    }
#pragma unroll
    for (int mt = 0; mt < 4; ++mt)
#pragma unroll
      for (int v = 0; v < 8; ++v)
        Sl[w][v + mh][mt * 16 + r16] = acc[mt][v] * invt;
    __syncthreads();
    if (lane < 16) {
      int node = nbase + lane;
      float mx = -1e30f;
      for (int m = 0; m < 64; ++m) mx = fmaxf(mx, Sl[w][lane][m]);
      float sum = 0.f;
      for (int m = 0; m < 64; ++m) sum += __expf(Sl[w][lane][m] - mx);
      float rs = 1.f / sum;
      if (node < NND) {
        for (int m = 0; m < 64; ++m) {
          float swv = __expf(Sl[w][lane][m] - mx) * rs;
          sw_nm[(long)node * 256 + h * 64 + m] = f2bf(swv);
          swT[((long)(h * 64 + m)) * NP + node] = f2bf(swv);
          atomicAdd(&nrm[h * 64 + m], swv);
        }
      }
    }
    __syncthreads();
  }
  for (int i = tid; i < 256; i += 128) atomicAdd(&normg[i], nrm[i]);
}

// ---------------- token reduction: tokacc[h][64][32] += swT_h @ fx_h (split-K over nodes) ----
__global__ __launch_bounds__(128) void k_tok(
    const __bf16* __restrict__ swT, const __bf16* __restrict__ fxT,
    float* __restrict__ tokacc) {
  int tid = threadIdx.x;
  int w = tid >> 5, lane = tid & 31;
  int r16 = lane & 15, kh = (lane >> 4) << 4, mh = (lane >> 4) << 3;
  int h = blockIdx.y;
  int mt = w;
  long kbase = (long)blockIdx.x * 2048;
  int ksteps = (int)((NP - kbase) / 32); if (ksteps > 64) ksteps = 64;
  f32x8 acc[2];
  acc[0] = zero8(); acc[1] = zero8();
  for (int ks = 0; ks < ksteps; ++ks) {
    long k = kbase + (long)ks * 32 + kh;
    bf16x16 a = *(const bf16x16*)(swT + ((long)(h * 64 + mt * 16 + r16)) * NP + k);
#pragma unroll
    for (int nt = 0; nt < 2; ++nt) {
      bf16x16 b = *(const bf16x16*)(fxT + ((long)(h * 32 + nt * 16 + r16)) * NP + k);
      acc[nt] = wmma_bf16(a, b, acc[nt]);
    }
  }
#pragma unroll
  for (int nt = 0; nt < 2; ++nt)
#pragma unroll
    for (int v = 0; v < 8; ++v)
      atomicAdd(&tokacc[(h * 64 + mt * 16 + v + mh) * 32 + nt * 16 + r16], acc[nt][v]);
}

// ---------------- tiny attention over tokens (one block: 256 threads = H*M rows) -------------
__global__ __launch_bounds__(256) void k_attn_small(
    const float* __restrict__ tokacc, const float* __restrict__ normg,
    const float* __restrict__ Wq, const float* __restrict__ Wk,
    const float* __restrict__ Wv, __bf16* __restrict__ otT) {
  __shared__ float kk2[256][32];
  __shared__ float vv[256][32];
  int t = threadIdx.x;          // t = h*64 + m
  int h = t >> 6;
  float tkr[32], qr[32];
  float nrmv = normg[t] + 1e-5f;
  for (int d = 0; d < 32; ++d) tkr[d] = tokacc[t * 32 + d] / nrmv;
  for (int d = 0; d < 32; ++d) {
    float sq = 0.f, sk = 0.f, sv = 0.f;
    for (int k = 0; k < 32; ++k) {
      float tv = tkr[k];
      sq += tv * Wq[k * 32 + d];
      sk += tv * Wk[k * 32 + d];
      sv += tv * Wv[k * 32 + d];
    }
    qr[d] = sq; kk2[t][d] = sk; vv[t][d] = sv;
  }
  __syncthreads();
  const float scale = 0.1767766953f; // 1/sqrt(32)
  float mx = -1e30f;
  for (int j = 0; j < 64; ++j) {
    float s = 0.f;
    for (int d = 0; d < 32; ++d) s += qr[d] * kk2[h * 64 + j][d];
    mx = fmaxf(mx, s * scale);
  }
  float sum = 0.f;
  for (int j = 0; j < 64; ++j) {
    float s = 0.f;
    for (int d = 0; d < 32; ++d) s += qr[d] * kk2[h * 64 + j][d];
    sum += __expf(s * scale - mx);
  }
  float rs = 1.f / sum;
  float o[32];
  for (int d = 0; d < 32; ++d) o[d] = 0.f;
  for (int j = 0; j < 64; ++j) {
    float s = 0.f;
    for (int d = 0; d < 32; ++d) s += qr[d] * kk2[h * 64 + j][d];
    float ww = __expf(s * scale - mx) * rs;
    for (int d = 0; d < 32; ++d) o[d] += ww * vv[h * 64 + j][d];
  }
  for (int d = 0; d < 32; ++d)
    otT[(h * 32 + d) * 64 + (t & 63)] = f2bf(o[d]);  // otT [H*32][64] (K=m contiguous)
}

// ---------------- de-slice (ox = sw @ ot) + Wo projection + residual ----------------
__global__ __launch_bounds__(128) void k_ox(
    const __bf16* __restrict__ sw_nm, const __bf16* __restrict__ otT_g,
    const __bf16* __restrict__ Wot, const float* __restrict__ bo,
    const float* __restrict__ V1, float* __restrict__ V2) {
  __shared__ __attribute__((aligned(32))) __bf16 Hl[64][128];
  __shared__ __attribute__((aligned(32))) __bf16 ot_s[128][64];
  int tid = threadIdx.x;
  int w = tid >> 5, lane = tid & 31;
  int r16 = lane & 15, kh = (lane >> 4) << 4, mh = (lane >> 4) << 3;
  for (int i = tid; i < 8192; i += 128) ((__bf16*)ot_s)[i] = otT_g[i];
  __syncthreads();
  int nbase = (blockIdx.x * 4 + w) * 16;
  int rowc = nbase + r16; if (rowc >= NND) rowc = NND - 1;
  for (int h = 0; h < 4; ++h) {
    f32x8 acc[2];
    acc[0] = zero8(); acc[1] = zero8();
#pragma unroll
    for (int kt = 0; kt < 2; ++kt) {
      bf16x16 a = *(const bf16x16*)(sw_nm + (long)rowc * 256 + h * 64 + kt * 32 + kh);
#pragma unroll
      for (int nt = 0; nt < 2; ++nt) {
        bf16x16 b = *(const bf16x16*)&ot_s[h * 32 + nt * 16 + r16][kt * 32 + kh];
        acc[nt] = wmma_bf16(a, b, acc[nt]);
      }
    }
#pragma unroll
    for (int nt = 0; nt < 2; ++nt)
#pragma unroll
      for (int v = 0; v < 8; ++v)
        Hl[w * 16 + v + mh][h * 32 + nt * 16 + r16] = f2bf(acc[nt][v]);
  }
  __syncthreads();
  f32x8 acc2[8];
#pragma unroll
  for (int t = 0; t < 8; ++t) acc2[t] = zero8();
#pragma unroll
  for (int kt = 0; kt < 4; ++kt) {
    bf16x16 a = *(const bf16x16*)&Hl[w * 16 + r16][kt * 32 + kh];
#pragma unroll
    for (int nt = 0; nt < 8; ++nt) {
      bf16x16 b = *(const bf16x16*)(Wot + (nt * 16 + r16) * 128 + kt * 32 + kh);
      acc2[nt] = wmma_bf16(a, b, acc2[nt]);
    }
  }
#pragma unroll
  for (int nt = 0; nt < 8; ++nt) {
    int n = nt * 16 + r16;
    float bb = bo[n];
#pragma unroll
    for (int v = 0; v < 8; ++v) {
      int orow = nbase + v + mh;
      if (orow < NND)
        V2[(long)orow * 128 + n] = V1[(long)orow * 128 + n] + acc2[nt][v] + bb;
    }
  }
}

// ---------------- fused FFN: GEMM(128->256) SiLU GEMM(256->128) + residual ----------------
__global__ __launch_bounds__(128) void k_ffn(
    const __bf16* __restrict__ h2, const __bf16* __restrict__ W1t,
    const float* __restrict__ b1, const __bf16* __restrict__ W2t,
    const float* __restrict__ b2, const float* __restrict__ V2,
    float* __restrict__ Vout) {
  __shared__ __attribute__((aligned(32))) __bf16 Hl[64][256];
  int tid = threadIdx.x;
  int w = tid >> 5, lane = tid & 31;
  int r16 = lane & 15, kh = (lane >> 4) << 4, mh = (lane >> 4) << 3;
  int nbase = (blockIdx.x * 4 + w) * 16;
  int rowc = nbase + r16; if (rowc >= NND) rowc = NND - 1;
#pragma unroll
  for (int half = 0; half < 2; ++half) {
    f32x8 acc[8];
#pragma unroll
    for (int t = 0; t < 8; ++t) acc[t] = zero8();
#pragma unroll
    for (int kt = 0; kt < 4; ++kt) {
      bf16x16 a = *(const bf16x16*)(h2 + (long)rowc * 128 + kt * 32 + kh);
#pragma unroll
      for (int nt = 0; nt < 8; ++nt) {
        bf16x16 b = *(const bf16x16*)(W1t + (half * 128 + nt * 16 + r16) * 128 + kt * 32 + kh);
        acc[nt] = wmma_bf16(a, b, acc[nt]);
      }
    }
#pragma unroll
    for (int nt = 0; nt < 8; ++nt) {
      int n = half * 128 + nt * 16 + r16;
      float bb = b1[n];
#pragma unroll
      for (int v = 0; v < 8; ++v)
        Hl[w * 16 + v + mh][n] = f2bf(silu(acc[nt][v] + bb));
    }
  }
  __syncthreads();
  f32x8 acc2[8];
#pragma unroll
  for (int t = 0; t < 8; ++t) acc2[t] = zero8();
#pragma unroll
  for (int kt = 0; kt < 8; ++kt) {
    bf16x16 a = *(const bf16x16*)&Hl[w * 16 + r16][kt * 32 + kh];
#pragma unroll
    for (int nt = 0; nt < 8; ++nt) {
      bf16x16 b = *(const bf16x16*)(W2t + (nt * 16 + r16) * 256 + kt * 32 + kh);
      acc2[nt] = wmma_bf16(a, b, acc2[nt]);
    }
  }
#pragma unroll
  for (int nt = 0; nt < 8; ++nt) {
    int n = nt * 16 + r16;
    float bb = b2[n];
#pragma unroll
    for (int v = 0; v < 8; ++v) {
      int orow = nbase + v + mh;
      if (orow < NND)
        Vout[(long)orow * 128 + n] = V2[(long)orow * 128 + n] + acc2[nt][v] + bb;
    }
  }
}

// ---------------- host launch ----------------
extern "C" void kernel_launch(void* const* d_in, const int* in_sizes, int n_in,
                              void* d_out, int out_size, void* d_ws, size_t ws_size,
                              hipStream_t stream) {
  (void)in_sizes; (void)n_in; (void)out_size; (void)ws_size;
  const float* V    = (const float*)d_in[0];
  const float* E    = (const float*)d_in[1];
  const int*   edg  = (const int*)d_in[2];
  const float* senc = (const float*)d_in[3];
  const float* cond = (const float*)d_in[4];
  const float* eW1  = (const float*)d_in[5];
  const float* eb1  = (const float*)d_in[6];
  const float* eW2  = (const float*)d_in[7];
  const float* eb2  = (const float*)d_in[8];
  const float* eg   = (const float*)d_in[9];
  const float* ebt  = (const float*)d_in[10];
  const float* nW1  = (const float*)d_in[11];
  const float* nb1  = (const float*)d_in[12];
  const float* nW2  = (const float*)d_in[13];
  const float* nb2  = (const float*)d_in[14];
  const float* ng   = (const float*)d_in[15];
  const float* nbt  = (const float*)d_in[16];
  const float* ln1g = (const float*)d_in[17];
  const float* ln1b = (const float*)d_in[18];
  const float* ln2g = (const float*)d_in[19];
  const float* ln2b = (const float*)d_in[20];
  const float* Wx   = (const float*)d_in[21];
  const float* bx   = (const float*)d_in[22];
  const float* Wfx  = (const float*)d_in[23];
  const float* bfx  = (const float*)d_in[24];
  const float* Wsl  = (const float*)d_in[25];
  const float* temp = (const float*)d_in[26];
  const float* Wq   = (const float*)d_in[27];
  const float* Wk   = (const float*)d_in[28];
  const float* Wv   = (const float*)d_in[29];
  const float* Wo   = (const float*)d_in[30];
  const float* bo   = (const float*)d_in[31];
  const float* fW1  = (const float*)d_in[32];
  const float* fb1  = (const float*)d_in[33];
  const float* fW2  = (const float*)d_in[34];
  const float* fb2  = (const float*)d_in[35];

  const int* src = edg;
  const int* dst = edg + NEDG;
  float* outV = (float*)d_out;
  float* outE = outV + (size_t)NND * 128;

  char* ws = (char*)d_ws;
  size_t off = 0;
  auto A = [&](size_t bytes) -> char* {
    char* p = ws + off; off += (bytes + 255) & ~(size_t)255; return p;
  };
  __bf16* eW1t = (__bf16*)A((size_t)128 * 448 * 2);
  __bf16* eW2t = (__bf16*)A((size_t)128 * 128 * 2);
  __bf16* nW1t = (__bf16*)A((size_t)128 * 288 * 2);
  __bf16* nW2t = (__bf16*)A((size_t)128 * 128 * 2);
  __bf16* Wxt  = (__bf16*)A((size_t)128 * 128 * 2);
  __bf16* Wfxt = (__bf16*)A((size_t)128 * 128 * 2);
  __bf16* WsT  = (__bf16*)A((size_t)64 * 32 * 2);
  __bf16* Wot  = (__bf16*)A((size_t)128 * 128 * 2);
  __bf16* fW1t = (__bf16*)A((size_t)256 * 128 * 2);
  __bf16* fW2t = (__bf16*)A((size_t)128 * 256 * 2);
  float*  Vp   = (float*)A((size_t)NND * 128 * 4);
  __bf16* Vin  = (__bf16*)A((size_t)NND * 160 * 2);
  __bf16* Ebf  = (__bf16*)A((size_t)NEDG * 128 * 2);
  float*  agg  = (float*)A((size_t)NND * 128 * 4);
  __bf16* aggbf= (__bf16*)A((size_t)NND * 128 * 2);
  float*  V1   = (float*)A((size_t)NND * 128 * 4);
  __bf16* xbf  = (__bf16*)A((size_t)NND * 128 * 2);
  __bf16* xm   = (__bf16*)A((size_t)NND * 128 * 2);
  __bf16* fxT  = (__bf16*)A((size_t)128 * NP * 2);
  __bf16* swnm = (__bf16*)A((size_t)NND * 256 * 2);
  __bf16* swT  = (__bf16*)A((size_t)256 * NP * 2);
  float*  normg= (float*)A((size_t)256 * 4);
  float*  tokac= (float*)A((size_t)8192 * 4);
  __bf16* otT  = (__bf16*)A((size_t)8192 * 2);
  float*  V2   = (float*)A((size_t)NND * 128 * 4);
  __bf16* h2   = (__bf16*)A((size_t)NND * 128 * 2);

  auto WT = [&](const float* Wsrc, __bf16* Wdst, int din, int dout) {
    int n = din * dout;
    k_w_t<<<(n + 255) / 256, 256, 0, stream>>>(Wsrc, Wdst, din, dout);
  };
  WT(eW1, eW1t, 448, 128); WT(eW2, eW2t, 128, 128);
  WT(nW1, nW1t, 288, 128); WT(nW2, nW2t, 128, 128);
  WT(Wx, Wxt, 128, 128);   WT(Wfx, Wfxt, 128, 128);
  WT(Wsl, WsT, 32, 64);    WT(Wo, Wot, 128, 128);
  WT(fW1, fW1t, 128, 256); WT(fW2, fW2t, 256, 128);

  hipMemsetAsync(agg,   0, (size_t)NND * 128 * 4, stream);
  hipMemsetAsync(normg, 0, (size_t)256 * 4, stream);
  hipMemsetAsync(tokac, 0, (size_t)8192 * 4, stream);
  hipMemsetAsync(fxT,   0, (size_t)128 * NP * 2, stream);
  hipMemsetAsync(swT,   0, (size_t)256 * NP * 2, stream);

  k_prep_nodes<<<(NND * 160 + 255) / 256, 256, 0, stream>>>(V, cond, senc, Vp, Vin);
  k_f2bf<<<(int)(((long)NEDG * 128 + 255) / 256), 256, 0, stream>>>(E, Ebf, (long)NEDG * 128);

  k_edge<<<NEDG / 64, 128, 0, stream>>>(Ebf, Vin, src, dst, eW1t, eb1, eW2t, eb2,
                                        eg, ebt, E, outE, agg);
  k_f2bf<<<(int)(((long)NND * 128 + 255) / 256), 256, 0, stream>>>(agg, aggbf, (long)NND * 128);
  k_node<<<(NND + 63) / 64, 128, 0, stream>>>(Vin, aggbf, nW1t, nb1, nW2t, nb2,
                                              ng, nbt, Vp, V1);

  k_ln_rows<<<(NND * 32 + 255) / 256, 256, 0, stream>>>(V1, ln1g, ln1b, xbf, NND);
  k_proj<<<(NND + 63) / 64, 128, 0, stream>>>(xbf, Wxt, Wfxt, bx, bfx, xm, fxT);
  k_slice<<<(NND + 63) / 64, 128, 0, stream>>>(xm, WsT, temp, swnm, swT, normg);
  k_tok<<<dim3(NP / 2048 + 1, 4), 128, 0, stream>>>(swT, fxT, tokac);
  k_attn_small<<<1, 256, 0, stream>>>(tokac, normg, Wq, Wk, Wv, otT);
  k_ox<<<(NND + 63) / 64, 128, 0, stream>>>(swnm, otT, Wot, bo, V1, V2);

  k_ln_rows<<<(NND * 32 + 255) / 256, 256, 0, stream>>>(V2, ln2g, ln2b, h2, NND);
  k_ffn<<<(NND + 63) / 64, 128, 0, stream>>>(h2, fW1t, fb1, fW2t, fb2, V2, outV);
}
